// clique_policy_net_7980049236416
// MI455X (gfx1250) — matrix operand order repaired and checked by reading, not compiled, not run on hardware
//
#include <hip/hip_runtime.h>
#include <hip/hip_bf16.h>

// ---------------- problem constants ----------------
#define NN   10000
#define KK   31
#define EE   (NN * KK)     // 310000 edges (= 19375 full 16-edge tiles)
#define NT   (EE / 16)     // 19375
#define SD   8
#define ID   2
#define ZD   32
#define ED   64
#define HH   128
#define FCC  128
#define AT   16
#define FT   6
#define DTC  0.25f

typedef __attribute__((ext_vector_type(16))) __bf16 v16bf;
typedef __attribute__((ext_vector_type(8)))  float  v8f;

#define DEVINL __device__ __forceinline__

DEVINL v8f wmma_bf16(v16bf a, v16bf b, v8f c) {
  // v_wmma_f32_16x16x32_bf16: D = A(16x32) * B(32x16) + C (f32 acc)
  return __builtin_amdgcn_wmma_f32_16x16x32_bf16(
      /*neg_a=*/false, a, /*neg_b=*/false, b,
      /*c_mod=*/(short)0, c, /*reuse_a=*/false, /*reuse_b=*/false);
}

// A fragment (16x32 bf16) from LDS, row-major, leading dim ld (elements).
// Lane L<16: row L, K = 0..15; lane L+16: row L, K = 16..31.
// Contiguous 8-element runs -> compiler emits 2x ds_load_b128.
DEVINL v16bf frag_a_lds(const __bf16* A, int ld, int lane) {
  const int half = (lane >> 4) & 1;
  const int r    = lane & 15;
  const __bf16* p = A + r * ld + half * 16;
  v16bf a;
#pragma unroll
  for (int i = 0; i < 8; ++i) a[i] = p[i];
#pragma unroll
  for (int i = 0; i < 8; ++i) a[8 + i] = p[8 + i];
  return a;
}

// B fragment from pre-swizzled buffer: [frag][lane][16] bf16, 32B/lane,
// fully coalesced -> 2x load_b128.
DEVINL v16bf frag_b_pk(const __bf16* P, int fragIdx, int lane) {
  return ((const v16bf*)P)[fragIdx * 32 + lane];
}

DEVINL float sigf(float x) { return 1.0f / (1.0f + __expf(-x)); }

union PK2 { uint32_t u; __bf16 h[2]; };
union PK4 { uint2 u; __bf16 h[4]; };
union PK8 { uint4 u; __bf16 h[8]; };

DEVINL uint2 pack4(float4 v) {
  PK4 p;
  p.h[0] = (__bf16)v.x; p.h[1] = (__bf16)v.y;
  p.h[2] = (__bf16)v.z; p.h[3] = (__bf16)v.w;
  return p.u;
}

// ---------------- weight pre-swizzle into WMMA-B fragment order ----------
// out[((q*nk + kk)*32 + lane)*16 + i] = W[k][c] with
//   k = kk*32 + (lane>>4)*16 + i   (row: B1 if k<rows1, B2 if k-off2 in [0,rows2))
//   c = q*16 + (lane&15)           (zero outside ncols)
__global__ __launch_bounds__(256) void k_pack(
    const float* __restrict__ B1, int rows1,
    const float* __restrict__ B2, int off2, int rows2,
    int ld, int ncols, int nk, int total, __bf16* __restrict__ out) {
  int idx = blockIdx.x * 256 + threadIdx.x;
  if (idx >= total) return;
  int f = idx >> 9;
  int q = f / nk, kk = f - q * nk;
  int r = (idx >> 4) & 31;
  int i = idx & 15;
  int k = kk * 32 + (r >> 4) * 16 + i;
  int c = q * 16 + (r & 15);
  float v = 0.0f;
  if (c < ncols) {
    if (k < rows1)                          v = B1[k * ld + c];
    else if (k >= off2 && k - off2 < rows2) v = B2[(k - off2) * ld + c];
  }
  out[idx] = (__bf16)v;
}

// ---------------- init: s0, s, u=0, LSTM h/c init GEMVs (K=8, VALU) -------
__global__ __launch_bounds__(256) void k_init(
    const float* __restrict__ hist,
    const float* __restrict__ oh0W, const float* __restrict__ oh0b,
    const float* __restrict__ oc0W, const float* __restrict__ oc0b,
    const float* __restrict__ sh0W, const float* __restrict__ sh0b,
    const float* __restrict__ sc0W, const float* __restrict__ sc0b,
    float* __restrict__ s, float* __restrict__ s0, float* __restrict__ u,
    float* __restrict__ oh, float* __restrict__ oc,
    float* __restrict__ sh, float* __restrict__ sc) {
  int idx = blockIdx.x * 256 + threadIdx.x;
  if (idx >= NN * HH) return;
  int n = idx / HH, h = idx % HH;
  const float* sr = hist + (size_t)3 * NN * SD + (size_t)n * SD;
  float sv[SD];
#pragma unroll
  for (int d = 0; d < SD; ++d) sv[d] = sr[d];
  float a = oh0b[h], b = oc0b[h], c = sh0b[h], e = sc0b[h];
#pragma unroll
  for (int d = 0; d < SD; ++d) {
    a += sv[d] * oh0W[d * HH + h];
    b += sv[d] * oc0W[d * HH + h];
    c += sv[d] * sh0W[d * HH + h];
    e += sv[d] * sc0W[d * HH + h];
  }
  oh[(size_t)n * HH + h] = a;
  oc[(size_t)n * HH + h] = b;
  sh[(size_t)n * HH + h] = c;
  sc[(size_t)n * HH + h] = e;
  if (h < SD) { s[n * SD + h] = sv[h]; s0[n * SD + h] = sv[h]; }
  if (h < ID) u[n * ID + h] = 0.0f;
}

// ---------------- edge encoding: enc = relu([s[i1],s[i2]] @ W_edge + b) ---
// 4 waves/block, one full 16-edge tile per wave (E is a multiple of 16).
__global__ __launch_bounds__(128) void k_edge(
    const float* __restrict__ s, const int* __restrict__ ei1,
    const int* __restrict__ ei2, const float* __restrict__ W_edge,
    const float* __restrict__ b_edge, __bf16* __restrict__ enc) {
  __shared__ __align__(16) __bf16 Wp[4 * 32 * 16];   // packed B frags, 4 KB
  __shared__ __align__(16) __bf16 Ash[4][16 * 32];   // 4 KB
  __shared__ __align__(16) __bf16 Csh[4][16 * ED];   // 8 KB output staging
  const int tid = threadIdx.x, lane = tid & 31, w = tid >> 5;

  // stage W_edge (K=16 padded to 32) in fragment order
  for (int i = tid; i < 4 * 512; i += 128) {
    int q = i >> 9, r = (i >> 4) & 31, ii = i & 15;
    int k = (r >> 4) * 16 + ii;
    int c = q * 16 + (r & 15);
    Wp[i] = (k < 16) ? (__bf16)W_edge[k * ED + c] : (__bf16)0.0f;
  }
  const int tile = blockIdx.x * 4 + w;
  const bool valid = (tile < NT);
  const long ebase = (long)tile * 16;
  if (valid) {
    int r = lane & 15, half = lane >> 4;
    long e = ebase + r;
    int node = (half ? ei2 : ei1)[e];
    const float4* s4 = (const float4*)s;
    float4 a0 = s4[(size_t)node * 2];
    float4 a1 = s4[(size_t)node * 2 + 1];
    PK8 p;
    p.h[0] = (__bf16)a0.x; p.h[1] = (__bf16)a0.y;
    p.h[2] = (__bf16)a0.z; p.h[3] = (__bf16)a0.w;
    p.h[4] = (__bf16)a1.x; p.h[5] = (__bf16)a1.y;
    p.h[6] = (__bf16)a1.z; p.h[7] = (__bf16)a1.w;
    *(uint4*)&Ash[w][r * 32 + half * 8] = p.u;
    uint4 zz = make_uint4(0, 0, 0, 0);
    *(uint4*)&Ash[w][r * 32 + 16 + half * 8] = zz;   // K pad
  }
  __syncthreads();

  if (valid) {
    v16bf a = frag_a_lds(&Ash[w][0], 32, lane);
    const int half = lane >> 4, cl = lane & 15;
#pragma unroll
    for (int q = 0; q < 4; ++q) {
      v8f acc = {};
      acc = wmma_bf16(a, frag_b_pk(Wp, q, lane), acc);
      int col = q * 16 + cl;
      float bias = b_edge[col];
#pragma unroll
      for (int j = 0; j < 8; ++j) {
        float v = acc[j] + bias;
        Csh[w][(j + 8 * half) * ED + col] = (__bf16)(v > 0.0f ? v : 0.0f);
      }
    }
  }
  __syncthreads();

  if (valid) {
    // whole 16x64 tile is contiguous 2 KB in row-major enc -> b128 copy
    const uint4* src = (const uint4*)&Csh[w][0];
    uint4* dst = (uint4*)(enc + (size_t)ebase * ED);
#pragma unroll
    for (int p = 0; p < 4; ++p) dst[p * 32 + lane] = src[p * 32 + lane];
  }
}

// ---------------- attention: scores (WMMA) + softmax + ctx ---------------
// 4 waves/block, one node per wave.
__global__ __launch_bounds__(128) void k_attn(
    const float* __restrict__ s, const float* __restrict__ s0,
    const int* __restrict__ nobs, const __bf16* __restrict__ enc,
    const float* __restrict__ att_W1, const float* __restrict__ att_W2,
    const float* __restrict__ att_v, float* __restrict__ ctx) {
  __shared__ __align__(16) __bf16 W1p[2 * 512];      // packed B frags, 2 KB
  __shared__ __align__(16) __bf16 Osh[4][32 * ED];   // 16 KB (row 31 zero)
  __shared__ float relW2sh[4][AT];
  __shared__ float scsh[4][32];
  const int tid = threadIdx.x, lane = tid & 31, w = tid >> 5;
  const int n = blockIdx.x * 4 + w;

  // stage att_W1 (64x16) in fragment order
  for (int i = tid; i < 2 * 512; i += 128) {
    int kk = i >> 9, r = (i >> 4) & 31, ii = i & 15;
    int k = kk * 32 + (r >> 4) * 16 + ii;
    int c = r & 15;
    W1p[i] = (__bf16)att_W1[k * AT + c];
  }
  if (lane < AT) {
    float acc = 0.0f;
#pragma unroll
    for (int d = 0; d < SD; ++d) {
      float rel = s[n * SD + d] - s0[n * SD + d];
      acc += rel * att_W2[d * AT + lane];
    }
    relW2sh[w][lane] = acc;
  }
  {
    const int half = lane >> 4, cl = lane & 15;
    for (int rr = 0; rr < 16; ++rr) {
      int r = rr * 2 + half;
      uint2 v = make_uint2(0, 0);
      if (r < KK) {
        int idx = nobs[n * KK + r];
        if (idx > 0)
          v = *(const uint2*)(enc + (size_t)(idx - 1) * ED + cl * 4);
      }
      *(uint2*)&Osh[w][r * ED + cl * 4] = v;   // r==31 -> zero pad row
    }
  }
  __syncthreads();

  const int half = lane >> 4;
  const float relv = relW2sh[w][lane & 15];
  const float vv   = att_v[lane & 15];
#pragma unroll
  for (int mt = 0; mt < 2; ++mt) {
    v8f acc = {};
#pragma unroll
    for (int kk = 0; kk < 2; ++kk) {
      v16bf a = frag_a_lds(&Osh[w][mt * 16 * ED + kk * 32], ED, lane);
      acc = wmma_bf16(a, frag_b_pk(W1p, kk, lane), acc);
    }
#pragma unroll
    for (int j = 0; j < 8; ++j) {
      float t = tanhf(acc[j] + relv) * vv;
      t += __shfl_xor(t, 1, 32);
      t += __shfl_xor(t, 2, 32);
      t += __shfl_xor(t, 4, 32);
      t += __shfl_xor(t, 8, 32);
      if ((lane & 15) == 0) scsh[w][mt * 16 + half * 8 + j] = t;
    }
  }
  __syncthreads();

  // softmax over 31 scores (lane 31 = -inf)
  float sval = (lane < KK) ? scsh[w][lane] : -1e30f;
  float m = sval;
#pragma unroll
  for (int o = 16; o >= 1; o >>= 1) m = fmaxf(m, __shfl_xor(m, o, 32));
  float e = (lane < KK) ? __expf(sval - m) : 0.0f;
  float ssum = e;
#pragma unroll
  for (int o = 16; o >= 1; o >>= 1) ssum += __shfl_xor(ssum, o, 32);
  scsh[w][lane] = e / ssum;
  __syncthreads();

  // ctx = alpha^T * obs, two adjacent columns per lane
  {
    float ax = 0.0f, ay = 0.0f;
    for (int r = 0; r < KK; ++r) {
      PK2 t;
      t.u = *(const uint32_t*)&Osh[w][r * ED + lane * 2];
      float al = scsh[w][r];
      ax += al * (float)t.h[0];
      ay += al * (float)t.h[1];
    }
    *(float2*)&ctx[(size_t)n * ED + lane * 2] = make_float2(ax, ay);
  }
}

// ---------------- fused per-16-node step: 2 LSTMs + action net + dyn -----
__global__ __launch_bounds__(128) void k_step(
    const float* __restrict__ s0g, const float* __restrict__ z,
    const float* __restrict__ ctx,
    const __bf16* __restrict__ obsP, const float* __restrict__ obs_b,
    const __bf16* __restrict__ stP,  const float* __restrict__ st_b,
    const __bf16* __restrict__ act1P, const float* __restrict__ act_b1,
    const __bf16* __restrict__ act2P, const float* __restrict__ act_b2,
    const float* __restrict__ dyn_A, const float* __restrict__ dyn_B,
    float* __restrict__ s, float* __restrict__ u,
    float* __restrict__ oh, float* __restrict__ oc,
    float* __restrict__ sh, float* __restrict__ sc,
    float* __restrict__ out_state, float* __restrict__ out_input) {
  __shared__ __align__(16) __bf16 Abuf[16 * 192]; // A_obs / A_st / hidden
  __shared__ __align__(16) __bf16 Aact[16 * 320]; // action-net A, 10 KB
  __shared__ __align__(16) float  G[16 * 512];    // gate tile, 32 KB
  __shared__ float u2sh[16 * ID];
  const int tid = threadIdx.x, lane = tid & 31, w = tid >> 5;
  const int nb = blockIdx.x * 16;
  const int half = lane >> 4;

  // P1: A_obs = [ctx | oh] (vectorized); static Aact parts: rel, z, pad
  {
    const float4* c4 = (const float4*)ctx;
    for (int i = tid; i < 16 * 16; i += 128) {       // ctx: 16 nodes x 16 f4
      int n = i >> 4, p = i & 15;
      *(uint2*)&Abuf[n * 192 + p * 4] = pack4(c4[(size_t)(nb + n) * 16 + p]);
    }
    const float4* o4 = (const float4*)oh;
    for (int i = tid; i < 16 * 32; i += 128) {       // oh: 16 nodes x 32 f4
      int n = i >> 5, p = i & 31;
      *(uint2*)&Abuf[n * 192 + 64 + p * 4] = pack4(o4[(size_t)(nb + n) * 32 + p]);
    }
    for (int i = tid; i < 16 * 64; i += 128) {       // Aact cols 0-7,264-319
      int n = i >> 6, c = i & 63;
      __bf16 v; int dst;
      if (c < SD)           { v = (__bf16)(s[(nb + n) * SD + c] - s0g[(nb + n) * SD + c]); dst = c; }
      else if (c < SD + ZD) { v = (__bf16)z[(size_t)(nb + n) * ZD + (c - SD)]; dst = 264 + (c - SD); }
      else                  { v = (__bf16)0.0f; dst = 296 + (c - SD - ZD); }
      Aact[n * 320 + dst] = v;
    }
  }
  __syncthreads();

  // P2: obs gates [16x512] = [16x192] @ [obs_Wi ; obs_Wh]
  {
    v16bf afr[6];
#pragma unroll
    for (int kk = 0; kk < 6; ++kk) afr[kk] = frag_a_lds(&Abuf[kk * 32], 192, lane);
#pragma unroll
    for (int qq = 0; qq < 8; ++qq) {
      int q = w * 8 + qq;
      v8f acc = {};
#pragma unroll
      for (int kk = 0; kk < 6; ++kk)
        acc = wmma_bf16(afr[kk], frag_b_pk(obsP, q * 6 + kk, lane), acc);
      int col = q * 16 + (lane & 15);
      float bias = obs_b[col];
#pragma unroll
      for (int j = 0; j < 8; ++j) G[(j + 8 * half) * 512 + col] = acc[j] + bias;
    }
  }
  __syncthreads();

  // P3: obs LSTM pointwise (2 h per iter) -> oh2, oc2; oh2 into Aact
  for (int i = tid; i < 16 * 64; i += 128) {
    int n = i >> 6, h2 = (i & 63) * 2;
    const float* Gr = &G[n * 512];
    float2 gi = *(const float2*)&Gr[h2];
    float2 gf = *(const float2*)&Gr[HH + h2];
    float2 gg = *(const float2*)&Gr[2 * HH + h2];
    float2 go = *(const float2*)&Gr[3 * HH + h2];
    float2 co = *(const float2*)&oc[(size_t)(nb + n) * HH + h2];
    float cx = sigf(gf.x) * co.x + sigf(gi.x) * tanhf(gg.x);
    float cy = sigf(gf.y) * co.y + sigf(gi.y) * tanhf(gg.y);
    float hx = sigf(go.x) * tanhf(cx);
    float hy = sigf(go.y) * tanhf(cy);
    *(float2*)&oc[(size_t)(nb + n) * HH + h2] = make_float2(cx, cy);
    *(float2*)&oh[(size_t)(nb + n) * HH + h2] = make_float2(hx, hy);
    PK2 p; p.h[0] = (__bf16)hx; p.h[1] = (__bf16)hy;
    *(uint32_t*)&Aact[n * 320 + SD + h2] = p.u;
  }
  __syncthreads();

  // P4: A_st = [rel(8) | u(2) | pad(6) | sh(128) | pad(16)]  (K pad -> 160)
  for (int i = tid; i < 16 * 16; i += 128) {
    int n = i >> 4, c = i & 15;
    float v = 0.0f;
    if (c < SD)           v = s[(nb + n) * SD + c] - s0g[(nb + n) * SD + c];
    else if (c < SD + ID) v = u[(nb + n) * ID + (c - SD)];
    Abuf[n * 160 + c] = (__bf16)v;
  }
  {
    const float4* h4 = (const float4*)sh;
    for (int i = tid; i < 16 * 32; i += 128) {
      int n = i >> 5, p = i & 31;
      *(uint2*)&Abuf[n * 160 + 16 + p * 4] = pack4(h4[(size_t)(nb + n) * 32 + p]);
    }
    for (int i = tid; i < 16 * 8; i += 128) {        // pad cols 144-159
      int n = i >> 3, p = i & 7;
      *(uint32_t*)&Abuf[n * 160 + 144 + p * 2] = 0u;
    }
  }
  __syncthreads();

  // P5: st gates [16x512] = [16x160] @ [st_Wi@0 ; st_Wh@16]
  {
    v16bf afr[5];
#pragma unroll
    for (int kk = 0; kk < 5; ++kk) afr[kk] = frag_a_lds(&Abuf[kk * 32], 160, lane);
#pragma unroll
    for (int qq = 0; qq < 8; ++qq) {
      int q = w * 8 + qq;
      v8f acc = {};
#pragma unroll
      for (int kk = 0; kk < 5; ++kk)
        acc = wmma_bf16(afr[kk], frag_b_pk(stP, q * 5 + kk, lane), acc);
      int col = q * 16 + (lane & 15);
      float bias = st_b[col];
#pragma unroll
      for (int j = 0; j < 8; ++j) G[(j + 8 * half) * 512 + col] = acc[j] + bias;
    }
  }
  __syncthreads();

  // P6: st LSTM pointwise -> sh2, sc2; sh2 into Aact
  for (int i = tid; i < 16 * 64; i += 128) {
    int n = i >> 6, h2 = (i & 63) * 2;
    const float* Gr = &G[n * 512];
    float2 gi = *(const float2*)&Gr[h2];
    float2 gf = *(const float2*)&Gr[HH + h2];
    float2 gg = *(const float2*)&Gr[2 * HH + h2];
    float2 go = *(const float2*)&Gr[3 * HH + h2];
    float2 co = *(const float2*)&sc[(size_t)(nb + n) * HH + h2];
    float cx = sigf(gf.x) * co.x + sigf(gi.x) * tanhf(gg.x);
    float cy = sigf(gf.y) * co.y + sigf(gi.y) * tanhf(gg.y);
    float hx = sigf(go.x) * tanhf(cx);
    float hy = sigf(go.y) * tanhf(cy);
    *(float2*)&sc[(size_t)(nb + n) * HH + h2] = make_float2(cx, cy);
    *(float2*)&sh[(size_t)(nb + n) * HH + h2] = make_float2(hx, hy);
    PK2 p; p.h[0] = (__bf16)hx; p.h[1] = (__bf16)hy;
    *(uint32_t*)&Aact[n * 320 + SD + HH + h2] = p.u;
  }
  __syncthreads();

  // P7: hidden = relu(Aact @ act_W1 + b1), K padded 296 -> 320
#pragma unroll
  for (int qq = 0; qq < 2; ++qq) {
    int q = w * 2 + qq;
    v8f acc = {};
#pragma unroll
    for (int kk = 0; kk < 10; ++kk) {
      v16bf a = frag_a_lds(&Aact[kk * 32], 320, lane);
      acc = wmma_bf16(a, frag_b_pk(act1P, q * 10 + kk, lane), acc);
    }
    int col = q * 16 + (lane & 15);
    float bias = act_b1[col];
#pragma unroll
    for (int j = 0; j < 8; ++j) {
      float v = acc[j] + bias;
      Abuf[(j + 8 * half) * FCC + col] = (__bf16)(v > 0.0f ? v : 0.0f);
    }
  }
  __syncthreads();

  // P8: u2 = hidden @ act_W2 + b2 (N=2, packer zero-pads cols), wave 0
  if (w == 0) {
    v8f acc = {};
#pragma unroll
    for (int kk = 0; kk < 4; ++kk) {
      v16bf a = frag_a_lds(&Abuf[kk * 32], FCC, lane);
      acc = wmma_bf16(a, frag_b_pk(act2P, kk, lane), acc);
    }
    int col = lane & 15;
    if (col < ID) {
      float bias = act_b2[col];
#pragma unroll
      for (int j = 0; j < 8; ++j) {
        int mm = j + 8 * half;
        float uv = acc[j] + bias;
        u2sh[mm * ID + col] = uv;
        u[(nb + mm) * ID + col] = uv;
        out_input[(size_t)(nb + mm) * ID + col] = uv;
      }
    }
  }
  __syncthreads();

  // P9: s2 = s + DT*(s @ dyn_A^T + u2 @ dyn_B^T)
  {
    int n = tid >> 3, d = tid & 7;
    float sv[SD];
#pragma unroll
    for (int j = 0; j < SD; ++j) sv[j] = s[(nb + n) * SD + j];
    float acc = 0.0f;
#pragma unroll
    for (int j = 0; j < SD; ++j) acc += sv[j] * dyn_A[d * SD + j];
    acc += u2sh[n * ID + 0] * dyn_B[d * ID + 0] + u2sh[n * ID + 1] * dyn_B[d * ID + 1];
    float s2 = sv[d] + DTC * acc;
    s[(nb + n) * SD + d] = s2;
    out_state[(size_t)(nb + n) * SD + d] = s2;
  }
}

// ---------------- host launcher ----------------
extern "C" void kernel_launch(void* const* d_in, const int* in_sizes, int n_in,
                              void* d_out, int out_size, void* d_ws, size_t ws_size,
                              hipStream_t stream) {
  const float* hist   = (const float*)d_in[0];
  const float* z      = (const float*)d_in[1];
  const int*   ei1    = (const int*)d_in[2];
  const int*   ei2    = (const int*)d_in[3];
  const int*   nobs   = (const int*)d_in[4];
  const float* W_edge = (const float*)d_in[5];
  const float* b_edge = (const float*)d_in[6];
  const float* att_W1 = (const float*)d_in[7];
  const float* att_W2 = (const float*)d_in[8];
  const float* att_v  = (const float*)d_in[9];
  const float* obs_Wi = (const float*)d_in[10];
  const float* obs_Wh = (const float*)d_in[11];
  const float* obs_b  = (const float*)d_in[12];
  const float* st_Wi  = (const float*)d_in[13];
  const float* st_Wh  = (const float*)d_in[14];
  const float* st_b   = (const float*)d_in[15];
  const float* oh0W   = (const float*)d_in[16];
  const float* oh0b   = (const float*)d_in[17];
  const float* oc0W   = (const float*)d_in[18];
  const float* oc0b   = (const float*)d_in[19];
  const float* sh0W   = (const float*)d_in[20];
  const float* sh0b   = (const float*)d_in[21];
  const float* sc0W   = (const float*)d_in[22];
  const float* sc0b   = (const float*)d_in[23];
  const float* act_W1 = (const float*)d_in[24];
  const float* act_b1 = (const float*)d_in[25];
  const float* act_W2 = (const float*)d_in[26];
  const float* act_b2 = (const float*)d_in[27];
  const float* dyn_A  = (const float*)d_in[28];
  const float* dyn_B  = (const float*)d_in[29];

  char* ws = (char*)d_ws;
  auto carve = [&](size_t bytes) -> char* {
    char* p = ws;
    ws += (bytes + 255) & ~(size_t)255;
    return p;
  };
  __bf16* enc   = (__bf16*)carve((size_t)EE * ED * sizeof(__bf16));
  float*  s     = (float*)carve((size_t)NN * SD * sizeof(float));
  float*  s0    = (float*)carve((size_t)NN * SD * sizeof(float));
  float*  u     = (float*)carve((size_t)NN * ID * sizeof(float));
  float*  ctx   = (float*)carve((size_t)NN * ED * sizeof(float));
  float*  oh    = (float*)carve((size_t)NN * HH * sizeof(float));
  float*  oc    = (float*)carve((size_t)NN * HH * sizeof(float));
  float*  sh    = (float*)carve((size_t)NN * HH * sizeof(float));
  float*  sc    = (float*)carve((size_t)NN * HH * sizeof(float));
  __bf16* obsP  = (__bf16*)carve((size_t)32 * 6 * 512 * sizeof(__bf16));
  __bf16* stP   = (__bf16*)carve((size_t)32 * 5 * 512 * sizeof(__bf16));
  __bf16* act1P = (__bf16*)carve((size_t)8 * 10 * 512 * sizeof(__bf16));
  __bf16* act2P = (__bf16*)carve((size_t)1 * 4 * 512 * sizeof(__bf16));

  float* out_state = (float*)d_out;                        // [FT, N, SD]
  float* out_input = (float*)d_out + (size_t)FT * NN * SD; // [FT, N, ID]

  // pre-swizzle weights into WMMA-B fragment order (bf16), once per call
  {
    int t;
    t = 32 * 6 * 512;
    k_pack<<<(t + 255) / 256, 256, 0, stream>>>(obs_Wi, 64, obs_Wh, 64, 128,
                                                4 * HH, 4 * HH, 6, t, obsP);
    t = 32 * 5 * 512;
    k_pack<<<(t + 255) / 256, 256, 0, stream>>>(st_Wi, 10, st_Wh, 16, 128,
                                                4 * HH, 4 * HH, 5, t, stP);
    t = 8 * 10 * 512;
    k_pack<<<(t + 255) / 256, 256, 0, stream>>>(act_W1, 296, nullptr, 0, 0,
                                                FCC, FCC, 10, t, act1P);
    t = 1 * 4 * 512;
    k_pack<<<(t + 255) / 256, 256, 0, stream>>>(act_W2, 128, nullptr, 0, 0,
                                                ID, ID, 4, t, act2P);
  }

  k_init<<<(NN * HH + 255) / 256, 256, 0, stream>>>(
      hist, oh0W, oh0b, oc0W, oc0b, sh0W, sh0b, sc0W, sc0b,
      s, s0, u, oh, oc, sh, sc);

  for (int t = 0; t < FT; ++t) {
    k_edge<<<(NT + 3) / 4, 128, 0, stream>>>(s, ei1, ei2, W_edge, b_edge, enc);
    k_attn<<<NN / 4, 128, 0, stream>>>(s, s0, nobs, enc, att_W1, att_W2, att_v, ctx);
    k_step<<<NN / 16, 128, 0, stream>>>(
        s0, z, ctx, obsP, obs_b, stP, st_b, act1P, act_b1, act2P, act_b2,
        dyn_A, dyn_B, s, u, oh, oc, sh, sc,
        out_state + (size_t)t * NN * SD, out_input + (size_t)t * NN * ID);
  }
}
